// RotomerSet2SetModel_41592463294499
// MI455X (gfx1250) — matrix operand order repaired and checked by reading, not compiled, not run on hardware
//
#include <hip/hip_runtime.h>
#include <hip/hip_fp16.h>
#include <math.h>

// ---------------------------------------------------------------------------
// Types for CDNA5 WMMA
// ---------------------------------------------------------------------------
typedef __attribute__((ext_vector_type(16))) _Float16 v16h;
typedef __attribute__((ext_vector_type(8)))  _Float16 v8h;
typedef __attribute__((ext_vector_type(8)))  float    v8f;

#define BATCH 32
#define NTOK  1024
#define TOK   (BATCH * NTOK)   // 32768
#define DMODEL 1024

// ---------------------------------------------------------------------------
// Wave (32-lane) reductions
// ---------------------------------------------------------------------------
__device__ __forceinline__ float waveReduceSum(float v) {
  #pragma unroll
  for (int off = 16; off > 0; off >>= 1) v += __shfl_xor(v, off, 32);
  return v;
}
__device__ __forceinline__ float waveReduceMax(float v) {
  #pragma unroll
  for (int off = 16; off > 0; off >>= 1) v = fmaxf(v, __shfl_xor(v, off, 32));
  return v;
}

// ---------------------------------------------------------------------------
// Fragment loader for one 32-wide K-chunk:
//   A (16x32 f16): lane(m=l16, half) holds K = {k0+half*16 ..+16}
//   B (32x16 f16) x4 tiles: lane(n=l16, ks=half) holds K = {k0+half*16 ..+16}
// ---------------------------------------------------------------------------
__device__ __forceinline__ void load_frags(const _Float16* __restrict__ arow,
                                           const _Float16* __restrict__ wrow0,
                                           int K, int k0, int half,
                                           v16h& av, v16h bv[4])
{
  av = *(const v16h*)(arow + k0 + half * 16);
  #pragma unroll
  for (int j = 0; j < 4; ++j)
    bv[j] = *(const v16h*)(wrow0 + (size_t)(j * 16) * K + k0 + half * 16);
}

// ---------------------------------------------------------------------------
// Generic WMMA GEMM:  D[M,N] = A[M,K] @ W[N,K]^T  (+bias | +Cin), f16 in / f32 acc
// Each wave computes a 16x64 output tile (4 N-tiles); A is loaded once per
// K-chunk and reused by 4 independent WMMAs.  K-chunks are double-buffered:
// loads for chunk k+1 are in flight while chunk k's 4 WMMAs execute, so after
// warm-up the matrix unit does not wait on fresh L2 loads.
// Requirements (guaranteed by call sites): M % 16 == 0, N % 64 == 0,
// K % 64 == 0 (even number of 32-chunks), wave-tile count % 4 == 0.
// No divergent guards around WMMA: EXEC is all-ones.
// ---------------------------------------------------------------------------
__global__ __launch_bounds__(128)
void wmma_gemm_f16(const _Float16* __restrict__ A, const _Float16* __restrict__ W,
                   const float* __restrict__ bias,   // nullable
                   const float* __restrict__ Cin,    // nullable (accumulate source)
                   float* __restrict__ Dout,
                   _Float16* __restrict__ D16,       // nullable f16 shadow output
                   int M, int N, int K)
{
  const int lane = threadIdx.x & 31;
  const int wave = threadIdx.x >> 5;
  const int tile = blockIdx.x * 4 + wave;     // 16x64 wave-tile index
  const int tilesN = N >> 6;                  // 64-wide tiles along N
  const int tm = tile / tilesN;
  const int tn = tile % tilesN;
  const int half = lane >> 4;   // 0 or 1 (half-wave select)
  const int l16  = lane & 15;

  const _Float16* __restrict__ arow  = A + (size_t)(tm * 16 + l16) * K;
  const _Float16* __restrict__ wrow0 = W + (size_t)(tn * 64 + l16) * K;

  v8f acc[4];
  #pragma unroll
  for (int j = 0; j < 4; ++j) {
    const int col = tn * 64 + j * 16 + l16;
    if (Cin) {
      #pragma unroll
      for (int v = 0; v < 8; ++v)
        acc[j][v] = Cin[(size_t)(tm * 16 + half * 8 + v) * N + col];
    } else if (bias) {
      const float bv = bias[col];
      #pragma unroll
      for (int v = 0; v < 8; ++v) acc[j][v] = bv;
    } else {
      #pragma unroll
      for (int v = 0; v < 8; ++v) acc[j][v] = 0.0f;
    }
  }

  const int kchunks = K >> 5;     // even by construction
  v16h av0, av1;
  v16h bv0[4], bv1[4];

  load_frags(arow, wrow0, K, 0, half, av0, bv0);          // warm-up: chunk 0
  for (int kc = 0; kc < kchunks; kc += 2) {
    // loads for chunk kc+1 overlap the 4 WMMAs of chunk kc
    load_frags(arow, wrow0, K, (kc + 1) << 5, half, av1, bv1);
    #pragma unroll
    for (int j = 0; j < 4; ++j)
      acc[j] = __builtin_amdgcn_wmma_f32_16x16x32_f16(false, av0, false, bv0[j],
                                                      (short)0, acc[j], false, false);
    // loads for chunk kc+2 overlap the 4 WMMAs of chunk kc+1
    if (kc + 2 < kchunks)
      load_frags(arow, wrow0, K, (kc + 2) << 5, half, av0, bv0);
    #pragma unroll
    for (int j = 0; j < 4; ++j)
      acc[j] = __builtin_amdgcn_wmma_f32_16x16x32_f16(false, av1, false, bv1[j],
                                                      (short)0, acc[j], false, false);
  }

  #pragma unroll
  for (int j = 0; j < 4; ++j) {
    const int col = tn * 64 + j * 16 + l16;
    #pragma unroll
    for (int v = 0; v < 8; ++v) {
      const size_t idx = (size_t)(tm * 16 + half * 8 + v) * N + col;
      const float x = acc[j][v];
      Dout[idx] = x;
      if (D16) D16[idx] = (_Float16)x;
    }
  }
}

// ---------------------------------------------------------------------------
// Setup kernels
// ---------------------------------------------------------------------------
__global__ void cvt_f16_kernel(const float* __restrict__ src, _Float16* __restrict__ dst, int n) {
  int i = blockIdx.x * blockDim.x + threadIdx.x;
  if (i < n) dst[i] = (_Float16)src[i];
}

// att1_w [128,2048] -> A-part (cols 0..1023) and B-part (cols 1024..2047), f16
__global__ void split_att1_kernel(const float* __restrict__ w,
                                  _Float16* __restrict__ a, _Float16* __restrict__ b) {
  int i = blockIdx.x * blockDim.x + threadIdx.x;   // 0 .. 128*1024-1
  int r = i >> 10, c = i & 1023;
  a[i] = (_Float16)w[r * 2048 + c];
  b[i] = (_Float16)w[r * 2048 + 1024 + c];
}

__global__ void bias_sum_kernel(const float* __restrict__ b0, const float* __restrict__ b1,
                                float* __restrict__ out, int n) {
  int i = blockIdx.x * blockDim.x + threadIdx.x;
  if (i < n) out[i] = b0[i] + b1[i];
}

// Embedding: emb[token, 256] = [amino(28) | elem(28) | pos(28) | relu(xyz@W+b)(172)]
__global__ __launch_bounds__(256)
void embed_kernel(const int* __restrict__ x_int, const float* __restrict__ x_xyz,
                  const float* __restrict__ amino, const float* __restrict__ elem,
                  const float* __restrict__ pos,
                  const float* __restrict__ xyz_w, const float* __restrict__ xyz_b,
                  _Float16* __restrict__ emb)
{
  const int t  = blockIdx.x;      // token
  const int ch = threadIdx.x;     // output channel
  const int* xi = x_int + t * 3;
  float v;
  if (ch < 28)       v = amino[xi[0] * 28 + ch];
  else if (ch < 56)  v = elem [xi[1] * 28 + (ch - 28)];
  else if (ch < 84)  v = pos  [xi[2] * 28 + (ch - 56)];
  else {
    const int o = ch - 84;        // 0..171
    const float* xw = xyz_w + o * 3;
    const float* xz = x_xyz + t * 3;
    v = fmaf(xz[0], xw[0], fmaf(xz[1], xw[1], fmaf(xz[2], xw[2], xyz_b[o])));
    v = fmaxf(v, 0.0f);
  }
  emb[(size_t)t * 256 + ch] = (_Float16)v;
}

// h=c=0; input_embed = [mean_n x[b,n,:], zeros]; f16 shadows
__global__ __launch_bounds__(1024)
void init_state_kernel(const float* __restrict__ x,
                       float* __restrict__ ie, _Float16* __restrict__ ie16,
                       float* __restrict__ h, _Float16* __restrict__ h16,
                       float* __restrict__ c)
{
  const int b = blockIdx.x, t = threadIdx.x;
  const float* xb = x + (size_t)b * NTOK * DMODEL + t;
  float s = 0.0f;
  for (int n = 0; n < NTOK; ++n) s += xb[(size_t)n * DMODEL];
  s *= (1.0f / (float)NTOK);
  ie  [b * 2048 + t]        = s;
  ie  [b * 2048 + 1024 + t] = 0.0f;
  ie16[b * 2048 + t]        = (_Float16)s;
  ie16[b * 2048 + 1024 + t] = (_Float16)0.0f;
  h  [b * 1024 + t] = 0.0f;
  h16[b * 1024 + t] = (_Float16)0.0f;
  c  [b * 1024 + t] = 0.0f;
}

// LSTM cell elementwise (PyTorch gate order i,f,g,o)
__global__ __launch_bounds__(256)
void lstm_kernel(const float* __restrict__ gates, float* __restrict__ c,
                 float* __restrict__ h, _Float16* __restrict__ h16)
{
  const int idx = blockIdx.x * blockDim.x + threadIdx.x;   // 0..32767
  const int b = idx >> 10, j = idx & 1023;
  const float ig = gates[b * 4096 + j];
  const float fg = gates[b * 4096 + 1024 + j];
  const float gg = gates[b * 4096 + 2048 + j];
  const float og = gates[b * 4096 + 3072 + j];
  const float si = 1.0f / (1.0f + expf(-ig));
  const float sf = 1.0f / (1.0f + expf(-fg));
  const float so = 1.0f / (1.0f + expf(-og));
  const float cc = sf * c[idx] + si * tanhf(gg);
  c[idx] = cc;
  const float hh = so * tanhf(cc);
  h[idx] = hh;
  h16[idx] = (_Float16)hh;
}

// logits[b,n] = att2_b + sum_k att2_w[k]*relu(xa[b,n,k] + ra[b,k] + att1_b[k])
__global__ __launch_bounds__(256)
void logits_kernel(const float* __restrict__ xa, const float* __restrict__ ra,
                   const float* __restrict__ att1_b, const float* __restrict__ att2_w,
                   const float* __restrict__ att2_b, float* __restrict__ logits)
{
  const int idx = blockIdx.x * blockDim.x + threadIdx.x;   // 0..32767
  const int b = idx >> 10;
  const float* __restrict__ xr = xa + (size_t)idx * 128;
  const float* __restrict__ rb = ra + b * 128;
  float s = att2_b[0];
  #pragma unroll 4
  for (int k = 0; k < 128; ++k) {
    const float a = xr[k] + rb[k] + att1_b[k];
    s += att2_w[k] * fmaxf(a, 0.0f);
  }
  logits[idx] = s;
}

// softmax over N per batch, pooled = sum_n p[n]*x[b,n,:], then pack next
// input_embed = [pooled | h] in f32 and f16
__global__ __launch_bounds__(1024)
void softmax_pool_kernel(const float* __restrict__ logits, const float* __restrict__ x,
                         const float* __restrict__ h,
                         float* __restrict__ ie, _Float16* __restrict__ ie16)
{
  __shared__ float p[NTOK];
  __shared__ float red[32];
  const int b = blockIdx.x, t = threadIdx.x;
  const int wave = t >> 5, lane = t & 31;

  const float l = logits[b * NTOK + t];
  float m = waveReduceMax(l);
  if (lane == 0) red[wave] = m;
  __syncthreads();
  if (wave == 0) { float mm = waveReduceMax(red[lane]); red[lane] = mm; }
  __syncthreads();
  m = red[0];
  __syncthreads();

  const float e = expf(l - m);
  p[t] = e;
  float s = waveReduceSum(e);
  if (lane == 0) red[wave] = s;
  __syncthreads();
  if (wave == 0) { float ss = waveReduceSum(red[lane]); red[lane] = ss; }
  __syncthreads();
  const float inv = 1.0f / red[0];   // p[*] also visible after this barrier

  float acc = 0.0f;
  const float* __restrict__ xb = x + (size_t)b * NTOK * DMODEL + t;
  for (int n = 0; n < NTOK; ++n) acc += p[n] * xb[(size_t)n * DMODEL];
  acc *= inv;

  const float hv = h[b * 1024 + t];
  ie  [b * 2048 + t]        = acc;
  ie  [b * 2048 + 1024 + t] = hv;
  ie16[b * 2048 + t]        = (_Float16)acc;
  ie16[b * 2048 + 1024 + t] = (_Float16)hv;
}

// out[b] = dot(c[b,:], energy_w) + energy_b
__global__ __launch_bounds__(256)
void energy_kernel(const float* __restrict__ c, const float* __restrict__ ew,
                   const float* __restrict__ eb, float* __restrict__ out)
{
  __shared__ float red[8];
  const int b = blockIdx.x, t = threadIdx.x;
  float s = 0.0f;
  for (int k = t; k < 1024; k += 256) s += c[b * 1024 + k] * ew[k];
  s = waveReduceSum(s);
  const int wave = t >> 5, lane = t & 31;
  if (lane == 0) red[wave] = s;
  __syncthreads();
  if (t == 0) {
    float tot = 0.0f;
    #pragma unroll
    for (int i = 0; i < 8; ++i) tot += red[i];
    out[b] = tot + eb[0];
  }
}

// ---------------------------------------------------------------------------
// Launcher
// ---------------------------------------------------------------------------
static inline char* ws_bump(char*& p, size_t bytes) {
  char* r = p;
  p += (bytes + 255) & ~(size_t)255;
  return r;
}

extern "C" void kernel_launch(void* const* d_in, const int* in_sizes, int n_in,
                              void* d_out, int out_size, void* d_ws, size_t ws_size,
                              hipStream_t stream) {
  (void)in_sizes; (void)n_in; (void)out_size; (void)ws_size;

  // Inputs (setup_inputs dict order)
  const int*   x_int      = (const int*)  d_in[0];   // [32,1024,3]
  const float* x_xyz      = (const float*)d_in[1];   // [32,1024,3]
  const float* amino      = (const float*)d_in[2];   // [20,28]
  const float* elem       = (const float*)d_in[3];   // [5,28]
  const float* pos        = (const float*)d_in[4];   // [21,28]
  const float* xyz_w      = (const float*)d_in[5];   // [172,3]
  const float* xyz_b      = (const float*)d_in[6];   // [172]
  const float* up_w       = (const float*)d_in[7];   // [1024,256]
  const float* up_b       = (const float*)d_in[8];   // [1024]
  const float* w_ih       = (const float*)d_in[9];   // [4096,2048]
  const float* w_hh       = (const float*)d_in[10];  // [4096,1024]
  const float* b_ih       = (const float*)d_in[11];  // [4096]
  const float* b_hh       = (const float*)d_in[12];  // [4096]
  const float* att1_w     = (const float*)d_in[13];  // [128,2048]
  const float* att1_b     = (const float*)d_in[14];  // [128]
  const float* att2_w     = (const float*)d_in[15];  // [1,128]
  const float* att2_b     = (const float*)d_in[16];  // [1]
  const float* energy_w   = (const float*)d_in[17];  // [1,1024]
  const float* energy_b   = (const float*)d_in[18];  // [1]
  float* out = (float*)d_out;                        // [32,1]

  // Workspace layout
  char* p = (char*)d_ws;
  float*    x_f32    = (float*)   ws_bump(p, (size_t)TOK * DMODEL * 4);   // 128 MB
  _Float16* x_f16    = (_Float16*)ws_bump(p, (size_t)TOK * DMODEL * 2);   //  64 MB
  _Float16* emb_f16  = (_Float16*)ws_bump(p, (size_t)TOK * 256 * 2);      //  16 MB
  float*    xa       = (float*)   ws_bump(p, (size_t)TOK * 128 * 4);      //  16 MB
  _Float16* up_w16   = (_Float16*)ws_bump(p, (size_t)1024 * 256 * 2);
  _Float16* w_ih16   = (_Float16*)ws_bump(p, (size_t)4096 * 2048 * 2);    //  16 MB
  _Float16* w_hh16   = (_Float16*)ws_bump(p, (size_t)4096 * 1024 * 2);    //   8 MB
  _Float16* att1A16  = (_Float16*)ws_bump(p, (size_t)128 * 1024 * 2);
  _Float16* att1B16  = (_Float16*)ws_bump(p, (size_t)128 * 1024 * 2);
  float*    bsum     = (float*)   ws_bump(p, 4096 * 4);
  float*    gates    = (float*)   ws_bump(p, (size_t)BATCH * 4096 * 4);
  float*    h_f32    = (float*)   ws_bump(p, (size_t)BATCH * 1024 * 4);
  _Float16* h_f16    = (_Float16*)ws_bump(p, (size_t)BATCH * 1024 * 2);
  float*    c_f32    = (float*)   ws_bump(p, (size_t)BATCH * 1024 * 4);
  float*    ra       = (float*)   ws_bump(p, (size_t)BATCH * 128 * 4);
  float*    logits   = (float*)   ws_bump(p, (size_t)BATCH * NTOK * 4);
  float*    ie_f32   = (float*)   ws_bump(p, (size_t)BATCH * 2048 * 4);
  _Float16* ie_f16   = (_Float16*)ws_bump(p, (size_t)BATCH * 2048 * 2);

  // ---- one-time weight conversion ----
  cvt_f16_kernel<<<(1024 * 256 + 255) / 256, 256, 0, stream>>>(up_w, up_w16, 1024 * 256);
  cvt_f16_kernel<<<(4096 * 2048 + 255) / 256, 256, 0, stream>>>(w_ih, w_ih16, 4096 * 2048);
  cvt_f16_kernel<<<(4096 * 1024 + 255) / 256, 256, 0, stream>>>(w_hh, w_hh16, 4096 * 1024);
  split_att1_kernel<<<(128 * 1024) / 256, 256, 0, stream>>>(att1_w, att1A16, att1B16);
  bias_sum_kernel<<<4096 / 256, 256, 0, stream>>>(b_ih, b_hh, bsum, 4096);

  // ---- embedding ----
  embed_kernel<<<TOK, 256, 0, stream>>>(x_int, x_xyz, amino, elem, pos, xyz_w, xyz_b, emb_f16);

  auto gemm = [&](const _Float16* A, const _Float16* W, const float* bias,
                  const float* Cin, float* D, _Float16* D16, int M, int N, int K) {
    const int wave_tiles = (M / 16) * (N / 64);   // 16x64 tile per wave
    wmma_gemm_f16<<<wave_tiles / 4, 128, 0, stream>>>(A, W, bias, Cin, D, D16, M, N, K);
  };

  // x = emb @ up_w^T + up_b        [32768,1024]  (f32 + f16 shadow)
  gemm(emb_f16, up_w16, up_b, nullptr, x_f32, x_f16, TOK, DMODEL, 256);

  // xa = x @ att1_w[:,1024:]^T     [32768,128]   (step-invariant -> hoisted)
  gemm(x_f16, att1B16, nullptr, nullptr, xa, nullptr, TOK, 128, DMODEL);

  // h=c=0; input_embed = [mean_n x, 0]
  init_state_kernel<<<BATCH, 1024, 0, stream>>>(x_f32, ie_f32, ie_f16, h_f32, h_f16, c_f32);

  const int STEPS = 6;
  for (int step = 0; step < STEPS; ++step) {
    // gates = ie @ w_ih^T + (b_ih + b_hh) + h @ w_hh^T
    gemm(ie_f16, w_ih16, bsum, nullptr, gates, nullptr, BATCH, 4096, 2048);
    gemm(h_f16,  w_hh16, nullptr, gates, gates, nullptr, BATCH, 4096, 1024);
    // LSTM cell -> c, h(read)
    lstm_kernel<<<(BATCH * 1024) / 256, 256, 0, stream>>>(gates, c_f32, h_f32, h_f16);

    if (step < STEPS - 1) {  // last step's attention result is never consumed
      // ra = read @ att1_w[:,:1024]^T   [32,128]
      gemm(h_f16, att1A16, nullptr, nullptr, ra, nullptr, BATCH, 128, DMODEL);
      logits_kernel<<<(BATCH * NTOK) / 256, 256, 0, stream>>>(xa, ra, att1_b, att2_w,
                                                              att2_b, logits);
      softmax_pool_kernel<<<BATCH, 1024, 0, stream>>>(logits, x_f32, h_f32, ie_f32, ie_f16);
    }
  }

  // out = c @ energy_w^T + energy_b
  energy_kernel<<<BATCH, 256, 0, stream>>>(c_f32, energy_w, energy_b, out);
}
